// GraphAttentionLayer_36490042147192
// MI455X (gfx1250) — matrix operand order repaired and checked by reading, not compiled, not run on hardware
//
#include <hip/hip_runtime.h>

// GAT layer, fused flash-style. N=8192, IN=512, OUT=256.
// d_in: [0]=h f32[8192*512], [1]=adj i32[8192*8192], [2]=W f32[512*256],
//       [3]=a_left f32[256], [4]=a_right f32[256].  d_out: f32[8192*256].

#define N_NODES 8192
#define IN_DIM  512
#define OUT_DIM 256
#define LEAKY   0.2f

typedef __attribute__((ext_vector_type(16))) __bf16 v16bf;
typedef __attribute__((ext_vector_type(8)))  __bf16 v8bf;
typedef __attribute__((ext_vector_type(8)))  float  v8f;

union BF16x16 { v16bf v; v8bf h[2]; };

// ---- async global -> LDS staging of one 16x32 int32 adj tile (2 KB) --------
// Executed by the 32 lanes of wave 0: 4 x global_load_async_to_lds_b128 per
// lane-set (128 x 16B chunks). Tracked by ASYNCcnt; DMA fills LDS while the
// other waves run WMMA.
__device__ __forceinline__ void issue_adj_tile(const int* __restrict__ adj,
                                               int rowbase, int j,
                                               int* ldsTile, int lane) {
#pragma unroll
  for (int i = 0; i < 4; ++i) {
    const int c   = i * 32 + lane;   // 16-byte chunk id, 0..127
    const int row = c >> 3;          // 8 chunks per 128B tile row
    const int co  = (c & 7) * 4;     // int offset within row
    const unsigned lds = (unsigned)(uintptr_t)(&ldsTile[row * 32 + co]); // low 32b = LDS offset
    const int* g = adj + (long)(rowbase + row) * N_NODES + j + co;
    asm volatile("global_load_async_to_lds_b128 %0, %1, off"
                 :: "v"(lds), "v"((unsigned long long)(uintptr_t)g)
                 : "memory");
  }
}
#define WAIT_ASYNC_LE4() asm volatile("s_wait_asynccnt 0x4" ::: "memory")
#define WAIT_ASYNC_0()   asm volatile("s_wait_asynccnt 0x0" ::: "memory")

// ---------------------------------------------------------------- kernel 0
// WT[n][k] = bf16(W[k][n])  (transpose W so GEMM1 B-fragments are contiguous)
__global__ void __launch_bounds__(256)
prep_wt(const float* __restrict__ W, __bf16* __restrict__ WT) {
  int idx = blockIdx.x * 256 + threadIdx.x;      // 512*256 elements
  int kk = idx >> 8, n = idx & 255;
  WT[n * IN_DIM + kk] = (__bf16)W[kk * OUT_DIM + n];
}

// ---------------------------------------------------------------- kernel 1
// WhT[n][i] = bf16( (h @ W)[i][n] ), via v_wmma_f32_16x16x32_bf16.
// Block = 4 waves; each wave owns 4 N-tiles (16 rows x 64 cols).
__global__ void __launch_bounds__(128)
gemm1_wht(const float* __restrict__ h, const __bf16* __restrict__ WT,
          __bf16* __restrict__ WhT) {
  const int rowbase = blockIdx.x * 16;
  const int wave = threadIdx.x >> 5;
  const int lane = threadIdx.x & 31;
  const int mrow = lane & 15;
  const int hi   = lane >> 4;                    // lane group (ISA A/B layouts)
  v8f acc[4] = {};
  const float* hrow = h + (long)(rowbase + mrow) * IN_DIM;

  for (int k = 0; k < IN_DIM; k += 32) {
    // A fragment (16-bit A 16x32): e0-7 -> K=k+hi*8+e ; e8-15 -> K=k+16+hi*8+e
    v8f f0 = *(const v8f*)(hrow + k + hi * 8);
    v8f f1 = *(const v8f*)(hrow + k + 16 + hi * 8);
    BF16x16 a;
    a.h[0] = __builtin_convertvector(f0, v8bf);
    a.h[1] = __builtin_convertvector(f1, v8bf);

    // batch all B loads (one clause, one wait), then 4 back-to-back WMMAs
    BF16x16 b[4];
#pragma unroll
    for (int t = 0; t < 4; ++t) {
      const int nb = (wave * 4 + t) * 16 + (lane & 15);   // N = lane&15
      const v8bf* bp = (const v8bf*)(WT + nb * IN_DIM + k + hi * 8);
      b[t].h[0] = bp[0]; b[t].h[1] = bp[2];
    }
#pragma unroll
    for (int t = 0; t < 4; ++t)
      acc[t] = __builtin_amdgcn_wmma_f32_16x16x32_bf16(
          false, a.v, false, b[t].v, (short)0, acc[t], false, false);
  }
  // D layout: VGPR r -> M = r + 8*hi, N = lane&15. Transposed bf16 store.
#pragma unroll
  for (int t = 0; t < 4; ++t) {
    const int nb = (wave * 4 + t) * 16 + (lane & 15);
#pragma unroll
    for (int r = 0; r < 8; ++r) {
      const int m = rowbase + r + 8 * hi;
      WhT[(long)nb * N_NODES + m] = (__bf16)acc[t][r];
    }
  }
}

// ---------------------------------------------------------------- kernel 2
// Wh1[i] = Wh[i]·a_left, Wh2[i] = Wh[i]·a_right  (reads bf16 WhT, coalesced)
__global__ void __launch_bounds__(256)
attn_vecs(const __bf16* __restrict__ WhT,
          const float* __restrict__ aL, const float* __restrict__ aR,
          float* __restrict__ Wh1, float* __restrict__ Wh2) {
  const int i = blockIdx.x * 256 + threadIdx.x;
  float s1 = 0.f, s2 = 0.f;
  for (int n = 0; n < OUT_DIM; ++n) {
    float w = (float)WhT[(long)n * N_NODES + i];
    s1 += w * aL[n];
    s2 += w * aR[n];
  }
  Wh1[i] = s1; Wh2[i] = s2;
}

// ---------------------------------------------------------------- kernel 3
// Pass A: per-row masked online softmax stats (m_i, 1/Z_i). One wave per row.
__global__ void __launch_bounds__(256)
softmax_stats(const int* __restrict__ adj, const float* __restrict__ Wh1,
              const float* __restrict__ Wh2,
              float* __restrict__ mOut, float* __restrict__ rZ) {
  const int wave = threadIdx.x >> 5, lane = threadIdx.x & 31;
  const int row  = blockIdx.x * 8 + wave;
  const float w1 = Wh1[row];
  const int4* arow = (const int4*)(adj + (long)row * N_NODES);
  float mL = -1e30f, sL = 0.f;
#define UPD(av, wv)                                              \
  if ((av) > 0) {                                                \
    float ev = w1 + (wv); ev = ev > 0.f ? ev : LEAKY * ev;       \
    if (ev > mL) { sL = sL * __expf(mL - ev) + 1.f; mL = ev; }   \
    else         { sL += __expf(ev - mL); }                      \
  }
  for (int it = 0; it < N_NODES / 128; ++it) {
    int4   a  = arow[it * 32 + lane];
    float4 w2 = ((const float4*)Wh2)[it * 32 + lane];
    UPD(a.x, w2.x) UPD(a.y, w2.y) UPD(a.z, w2.z) UPD(a.w, w2.w)
  }
#undef UPD
  for (int off = 16; off; off >>= 1) {                 // wave32 reduction
    float mo = __shfl_xor(mL, off, 32);
    float so = __shfl_xor(sL, off, 32);
    float mn = fmaxf(mL, mo);
    sL = sL * __expf(mL - mn) + so * __expf(mo - mn);
    mL = mn;
  }
  if (lane == 0) {
    if (sL > 0.f) { mOut[row] = mL;  rZ[row] = 1.f / sL; }
    else          { mOut[row] = 0.f; rZ[row] = 0.f; }    // isolated node
  }
}

// ---------------------------------------------------------------- kernel 4
// Pass B: h_out = softmax(e) @ Wh. adj tiles streamed to LDS by the async DMA
// (triple buffered, ASYNCcnt), P tile built in LDS (double buffered), consumed
// by 8 waves via WMMA. Fused 1/Z scale + ELU epilogue. One barrier per tile.
__global__ void __launch_bounds__(256)
attn_out(const int* __restrict__ adj, const __bf16* __restrict__ WhT,
         const float* __restrict__ Wh1, const float* __restrict__ Wh2,
         const float* __restrict__ mRow, const float* __restrict__ rZ,
         float* __restrict__ out) {
  __shared__ __align__(16) int    adjT[3][16][32];   // async DMA destination
  __shared__ __align__(16) __bf16 P[2][16][32];
  __shared__ float sWh1[16], sM[16], sRZ[16];
  const int rowbase = blockIdx.x * 16;
  const int tid  = threadIdx.x;
  const int wave = tid >> 5, lane = tid & 31;
  const int hi   = lane >> 4;
  if (tid < 16) { sWh1[tid] = Wh1[rowbase + tid];
                  sM[tid]   = mRow[rowbase + tid];
                  sRZ[tid]  = rZ[rowbase + tid]; }
  // prologue: two adj tiles in flight, guarantee tile 0 has landed
  if (wave == 0) {
    issue_adj_tile(adj, rowbase, 0,  &adjT[0][0][0], lane);
    issue_adj_tile(adj, rowbase, 32, &adjT[1][0][0], lane);
    WAIT_ASYNC_LE4();
  }
  __syncthreads();

  const int m0 = tid >> 5;          // P-builder: this thread fills rows m0, m0+8
  const int kc = tid & 31;          // ... at column kc
  v8f acc0 = {}, acc1 = {};
  const int nb0 = wave * 32 + (lane & 15);  // output cols for this wave's 2 tiles
  const int nb1 = nb0 + 16;

  for (int j = 0, it = 0; j < N_NODES; j += 32, ++it) {
    const int slot = it % 3;        // adjT[slot] holds tile j (landed + barrier)
    const int pp   = it & 1;
    if (wave == 0) {                // keep 2 tiles in flight, retire oldest
      if (j + 64 < N_NODES) {
        issue_adj_tile(adj, rowbase, j + 64, &adjT[(slot + 2) % 3][0][0], lane);
        WAIT_ASYNC_LE4();           // tile j+32 complete before this iter's barrier
      } else if (j + 32 < N_NODES) {
        WAIT_ASYNC_0();
      }
    }
    // ---- cooperative P tile (16x32 bf16): adj from LDS, exp, mask
    {
      const int a0 = adjT[slot][m0][kc];
      const int a1 = adjT[slot][m0 + 8][kc];
      const float w2 = Wh2[j + kc];
      float e0 = sWh1[m0]     + w2; e0 = e0 > 0.f ? e0 : LEAKY * e0;
      float e1 = sWh1[m0 + 8] + w2; e1 = e1 > 0.f ? e1 : LEAKY * e1;
      P[pp][m0][kc]     = (a0 > 0) ? (__bf16)__expf(e0 - sM[m0])     : (__bf16)0.f;
      P[pp][m0 + 8][kc] = (a1 > 0) ? (__bf16)__expf(e1 - sM[m0 + 8]) : (__bf16)0.f;
    }
    // ---- B fragments from bf16 WhT (issue before barrier to hide latency)
    BF16x16 b0, b1;
    const __bf16* wp0 = WhT + (long)nb0 * N_NODES + j + hi * 8;
    b0.h[0] = *(const v8bf*)wp0;       b0.h[1] = *(const v8bf*)(wp0 + 16);
    const __bf16* wp1 = WhT + (long)nb1 * N_NODES + j + hi * 8;
    b1.h[0] = *(const v8bf*)wp1;       b1.h[1] = *(const v8bf*)(wp1 + 16);
    __syncthreads();
    // ---- A fragment from LDS (ISA 16-bit A 16x32 layout)
    BF16x16 a;
    a.h[0] = *(const v8bf*)&P[pp][lane & 15][hi * 8];
    a.h[1] = *(const v8bf*)&P[pp][lane & 15][hi * 8 + 16];
    acc0 = __builtin_amdgcn_wmma_f32_16x16x32_bf16(
        false, a.v, false, b0.v, (short)0, acc0, false, false);
    acc1 = __builtin_amdgcn_wmma_f32_16x16x32_bf16(
        false, a.v, false, b1.v, (short)0, acc1, false, false);
  }
  // ---- epilogue: scale by 1/Z, ELU, coalesced f32 stores
#pragma unroll
  for (int r = 0; r < 8; ++r) {
    const int m = r + 8 * hi;
    const float z = sRZ[m];
    float x0 = acc0[r] * z, x1 = acc1[r] * z;
    x0 = x0 > 0.f ? x0 : (__expf(x0) - 1.f);
    x1 = x1 > 0.f ? x1 : (__expf(x1) - 1.f);
    out[(long)(rowbase + m) * OUT_DIM + nb0] = x0;
    out[(long)(rowbase + m) * OUT_DIM + nb1] = x1;
  }
}

// ---------------------------------------------------------------- launcher
extern "C" void kernel_launch(void* const* d_in, const int* in_sizes, int n_in,
                              void* d_out, int out_size, void* d_ws, size_t ws_size,
                              hipStream_t stream) {
  const float* h   = (const float*)d_in[0];
  const int*   adj = (const int*)  d_in[1];
  const float* W   = (const float*)d_in[2];
  const float* aL  = (const float*)d_in[3];
  const float* aR  = (const float*)d_in[4];
  float* out = (float*)d_out;

  // workspace carve-up (~4.5 MB total)
  char* ws = (char*)d_ws;
  __bf16* WT  = (__bf16*)(ws);                    // 256 KB
  __bf16* WhT = (__bf16*)(ws + (1u << 19));       // 4 MB
  float* Wh1 = (float*)(ws + (1u << 19) + (1u << 22));            // 32 KB each
  float* Wh2 = Wh1 + N_NODES;
  float* mR  = Wh2 + N_NODES;
  float* rZ  = mR  + N_NODES;

  prep_wt      <<<(IN_DIM * OUT_DIM) / 256, 256, 0, stream>>>(W, WT);
  gemm1_wht    <<<N_NODES / 16, 128, 0, stream>>>(h, WT, WhT);
  attn_vecs    <<<N_NODES / 256, 256, 0, stream>>>(WhT, aL, aR, Wh1, Wh2);
  softmax_stats<<<N_NODES / 8,   256, 0, stream>>>(adj, Wh1, Wh2, mR, rZ);
  attn_out     <<<N_NODES / 16,  256, 0, stream>>>(adj, WhT, Wh1, Wh2, mR, rZ, out);
}
